// GATNet_3375844295349
// MI455X (gfx1250) — matrix operand order repaired and checked by reading, not compiled, not run on hardware
//
#include <hip/hip_runtime.h>
#include <hip/hip_bf16.h>

typedef __attribute__((ext_vector_type(16))) _Float16 v16h;
typedef __attribute__((ext_vector_type(8)))  float    v8f;

#define NEG_SLOPE 0.2f
#define EPS_DEN   1e-16f
#define ORD_NEG_INF 0x007FFFFFu   // f2ord(-inf)

// ---------- helpers ----------
// order-preserving float<->uint for atomic max
__device__ __forceinline__ unsigned f2ord(float f) {
  unsigned u = __float_as_uint(f);
  return (u & 0x80000000u) ? ~u : (u | 0x80000000u);
}
__device__ __forceinline__ float ord2f(unsigned u) {
  return (u & 0x80000000u) ? __uint_as_float(u & 0x7FFFFFFFu)
                           : __uint_as_float(~u);
}
__device__ __forceinline__ float lrelu(float v) { return v > 0.f ? v : NEG_SLOPE * v; }

// ---------- W1 -> f16, packed into 16-halfword granules:
// element (col n, k) -> w1t[((k>>4)*64 + n)*16 + (k&15)]
// One granule = one B-fragment slice (col n, K = 16q..16q+15), 32B aligned.
__global__ void k_w1t(const float* __restrict__ W1, _Float16* __restrict__ w1t) {
  int i = blockIdx.x * blockDim.x + threadIdx.x;   // over 512*64
  if (i >= 512 * 64) return;
  int k = i >> 6, n = i & 63;
  w1t[(size_t)((k >> 4) * 64 + n) * 16 + (k & 15)] = (_Float16)W1[i];
}

// ---------- Layer-1 GEMM: h1[N,64] = x[N,512] @ W1  (f16 WMMA, f32 accum) ----------
// W1^T (64KB f16) staged in LDS once per block; one wave owns a 16-row stripe
// across all 64 output columns: 1 A-frag (HBM) feeds 4 WMMAs per k-step.
__global__ void __launch_bounds__(256) k_gemm1(const float* __restrict__ x,
                                               const _Float16* __restrict__ w1t,
                                               float* __restrict__ h1, int N) {
  __shared__ _Float16 lw[32 * 64 * 16];            // 65536 B = full W1^T in granule layout

  // cooperative linear 64KB copy: 256 threads x 16 x uint4
  {
    const uint4* src = reinterpret_cast<const uint4*>(w1t);
    uint4* dst = reinterpret_cast<uint4*>(lw);
    #pragma unroll
    for (int i = 0; i < 16; ++i)
      dst[threadIdx.x + 256 * i] = src[threadIdx.x + 256 * i];
  }
  __syncthreads();

  const int lane = threadIdx.x & 31;
  const int wave = threadIdx.x >> 5;
  const bool lo  = lane < 16;
  const int l15  = lane & 15;
  const int m0   = blockIdx.x * 128 + wave * 16;
  const int row  = m0 + l15;
  const int rowc = row < N ? row : (N - 1);        // clamp: loads always valid, stores guarded
  const int abase = lo ? 0 : 16;                   // A: lanes 0-15 -> K 0..15, hi -> 16..31
  const int kbadd = lo ? 0 : 1;                    // B: lanes 0-15 -> K 0..15, hi -> 16..31 (granule +1)
  const float* xr = x + (size_t)rowc * 512;
  const v16h* lwg = reinterpret_cast<const v16h*>(lw);   // granule view: [kblk*64 + col]

  v8f acc0 = {}, acc1 = {}, acc2 = {}, acc3 = {};
  for (int k0 = 0; k0 < 512; k0 += 32) {
    const float4* p0 = reinterpret_cast<const float4*>(xr + k0 + abase);
    const float4* p1 = reinterpret_cast<const float4*>(xr + k0 + abase + 8);
    float4 f0 = p0[0], f1 = p0[1], f2 = p1[0], f3 = p1[1];
    v16h a;
    a[0]=(_Float16)f0.x; a[1]=(_Float16)f0.y; a[2]=(_Float16)f0.z; a[3]=(_Float16)f0.w;
    a[4]=(_Float16)f1.x; a[5]=(_Float16)f1.y; a[6]=(_Float16)f1.z; a[7]=(_Float16)f1.w;
    a[8]=(_Float16)f2.x; a[9]=(_Float16)f2.y; a[10]=(_Float16)f2.z; a[11]=(_Float16)f2.w;
    a[12]=(_Float16)f3.x; a[13]=(_Float16)f3.y; a[14]=(_Float16)f3.z; a[15]=(_Float16)f3.w;

    const v16h* lb = lwg + ((k0 >> 4) + kbadd) * 64;   // this k-granule row, 64 cols
    v16h b0 = lb[l15];            // cols  0..15
    v16h b1 = lb[16 + l15];       // cols 16..31
    v16h b2 = lb[32 + l15];       // cols 32..47
    v16h b3 = lb[48 + l15];       // cols 48..63

    acc0 = __builtin_amdgcn_wmma_f32_16x16x32_f16(false, a, false, b0, (short)0, acc0, false, false);
    acc1 = __builtin_amdgcn_wmma_f32_16x16x32_f16(false, a, false, b1, (short)0, acc1, false, false);
    acc2 = __builtin_amdgcn_wmma_f32_16x16x32_f16(false, a, false, b2, (short)0, acc2, false, false);
    acc3 = __builtin_amdgcn_wmma_f32_16x16x32_f16(false, a, false, b3, (short)0, acc3, false, false);
  }
  // C layout: VGPR r -> M = r (lanes 0-15) / 8+r (lanes 16-31); N = lane&15
  #pragma unroll
  for (int r = 0; r < 8; ++r) {
    int rr = m0 + r + (lo ? 0 : 8);
    if (rr < N) {
      float* o = h1 + (size_t)rr * 64 + l15;
      o[0]  = acc0[r];
      o[16] = acc1[r];
      o[32] = acc2[r];
      o[48] = acc3[r];
    }
  }
}

// ---------- per-node attention logits, layer 1 (H=8, C=8) ----------
__global__ void k_alpha1(const float* __restrict__ h1, const float* __restrict__ aw_s,
                         const float* __restrict__ aw_d, float* __restrict__ as1,
                         float* __restrict__ ad1, long long N) {
  long long i = (long long)blockIdx.x * blockDim.x + threadIdx.x;  // over N*8
  if (i >= N * 8) return;
  long long n = i >> 3; int h = (int)(i & 7);
  const float* hp = h1 + n * 64 + h * 8;
  float s = 0.f, d = 0.f;
  #pragma unroll
  for (int c = 0; c < 8; ++c) { s += hp[c] * aw_s[h * 8 + c]; d += hp[c] * aw_d[h * 8 + c]; }
  as1[i] = s; ad1[i] = d;
}

__global__ void k_init1(float* agg, float* den, unsigned* m, long long N) {
  long long i = (long long)blockIdx.x * blockDim.x + threadIdx.x;  // over N*64
  if (i >= N * 64) return;
  agg[i] = 0.f;
  if (i < N * 8) { den[i] = 0.f; m[i] = ORD_NEG_INF; }
}

// ---------- edge passes, layer 1 ----------
__global__ void k_edge_max1(const long long* __restrict__ ei, long long E, long long N,
                            const float* __restrict__ as1, const float* __restrict__ ad1,
                            unsigned* __restrict__ m) {
  long long t = (long long)blockIdx.x * blockDim.x + threadIdx.x;  // (E+N)*8
  long long EE = E + N;
  if (t >= EE * 8) return;
  long long e = t >> 3; int h = (int)(t & 7);
  long long s, d;
  if (e < E) { s = ei[e]; d = ei[E + e]; } else { s = d = e - E; }
  float ev = lrelu(as1[s * 8 + h] + ad1[d * 8 + h]);
  atomicMax(&m[d * 8 + h], f2ord(ev));
}

__global__ void k_edge_sum1(const long long* __restrict__ ei, long long E, long long N,
                            const float* __restrict__ as1, const float* __restrict__ ad1,
                            const unsigned* __restrict__ m, float* __restrict__ den) {
  long long t = (long long)blockIdx.x * blockDim.x + threadIdx.x;
  long long EE = E + N;
  if (t >= EE * 8) return;
  long long e = t >> 3; int h = (int)(t & 7);
  long long s, d;
  if (e < E) { s = ei[e]; d = ei[E + e]; } else { s = d = e - E; }
  float ev = lrelu(as1[s * 8 + h] + ad1[d * 8 + h]);
  atomicAdd(&den[d * 8 + h], __expf(ev - ord2f(m[d * 8 + h])));
}

__global__ void k_edge_agg1(const long long* __restrict__ ei, long long E, long long N,
                            const float* __restrict__ as1, const float* __restrict__ ad1,
                            const unsigned* __restrict__ m, const float* __restrict__ den,
                            const float* __restrict__ h1, float* __restrict__ agg) {
  long long t = (long long)blockIdx.x * blockDim.x + threadIdx.x;  // (E+N)*64
  long long EE = E + N;
  if (t >= EE * 64) return;
  long long e = t >> 6; int f = (int)(t & 63); int h = f >> 3;
  long long s, d;
  if (e < E) { s = ei[e]; d = ei[E + e]; } else { s = d = e - E; }
  float ev = lrelu(as1[s * 8 + h] + ad1[d * 8 + h]);
  float p  = __expf(ev - ord2f(m[d * 8 + h]));
  float alpha = p / (den[d * 8 + h] + EPS_DEN);
  atomicAdd(&agg[d * 64 + f], alpha * h1[s * 64 + f]);
}

// ---------- bias + ELU (in place on agg1) ----------
__global__ void k_bias_elu(float* __restrict__ agg, const float* __restrict__ b1, long long N) {
  long long i = (long long)blockIdx.x * blockDim.x + threadIdx.x;  // N*64
  if (i >= N * 64) return;
  float v = agg[i] + b1[i & 63];
  agg[i] = v > 0.f ? v : (__expf(v) - 1.f);
}

// ---------- Layer-2 GEMM (K=64, 10 classes) ----------
__global__ void k_gemm2(const float* __restrict__ h2in, const float* __restrict__ W2,
                        float* __restrict__ h2, long long N) {
  long long i = (long long)blockIdx.x * blockDim.x + threadIdx.x;  // N*10
  if (i >= N * 10) return;
  long long n = i / 10; int c = (int)(i % 10);
  const float* hp = h2in + n * 64;
  float s = 0.f;
  #pragma unroll
  for (int k = 0; k < 64; ++k) s += hp[k] * W2[k * 10 + c];
  h2[i] = s;
}

__global__ void k_alpha2(const float* __restrict__ h2, const float* __restrict__ aw_s,
                         const float* __restrict__ aw_d, float* __restrict__ as2,
                         float* __restrict__ ad2, long long N) {
  long long n = (long long)blockIdx.x * blockDim.x + threadIdx.x;
  if (n >= N) return;
  const float* hp = h2 + n * 10;
  float s = 0.f, d = 0.f;
  #pragma unroll
  for (int c = 0; c < 10; ++c) { s += hp[c] * aw_s[c]; d += hp[c] * aw_d[c]; }
  as2[n] = s; ad2[n] = d;
}

__global__ void k_init2(float* agg, float* den, unsigned* m, long long N) {
  long long i = (long long)blockIdx.x * blockDim.x + threadIdx.x;  // N*10
  if (i >= N * 10) return;
  agg[i] = 0.f;
  if (i < N) { den[i] = 0.f; m[i] = ORD_NEG_INF; }
}

__global__ void k_edge_max2(const long long* __restrict__ ei, long long E, long long N,
                            const float* __restrict__ as2, const float* __restrict__ ad2,
                            unsigned* __restrict__ m) {
  long long e = (long long)blockIdx.x * blockDim.x + threadIdx.x;
  if (e >= E + N) return;
  long long s, d;
  if (e < E) { s = ei[e]; d = ei[E + e]; } else { s = d = e - E; }
  atomicMax(&m[d], f2ord(lrelu(as2[s] + ad2[d])));
}

__global__ void k_edge_sum2(const long long* __restrict__ ei, long long E, long long N,
                            const float* __restrict__ as2, const float* __restrict__ ad2,
                            const unsigned* __restrict__ m, float* __restrict__ den) {
  long long e = (long long)blockIdx.x * blockDim.x + threadIdx.x;
  if (e >= E + N) return;
  long long s, d;
  if (e < E) { s = ei[e]; d = ei[E + e]; } else { s = d = e - E; }
  atomicAdd(&den[d], __expf(lrelu(as2[s] + ad2[d]) - ord2f(m[d])));
}

__global__ void k_edge_agg2(const long long* __restrict__ ei, long long E, long long N,
                            const float* __restrict__ as2, const float* __restrict__ ad2,
                            const unsigned* __restrict__ m, const float* __restrict__ den,
                            const float* __restrict__ h2, float* __restrict__ agg) {
  long long e = (long long)blockIdx.x * blockDim.x + threadIdx.x;
  if (e >= E + N) return;
  long long s, d;
  if (e < E) { s = ei[e]; d = ei[E + e]; } else { s = d = e - E; }
  float p = __expf(lrelu(as2[s] + ad2[d]) - ord2f(m[d]));
  float alpha = p / (den[d] + EPS_DEN);
  #pragma unroll
  for (int c = 0; c < 10; ++c) atomicAdd(&agg[d * 10 + c], alpha * h2[s * 10 + c]);
}

__global__ void k_logsm(const float* __restrict__ agg, const float* __restrict__ b2,
                        float* __restrict__ out, long long N) {
  long long n = (long long)blockIdx.x * blockDim.x + threadIdx.x;
  if (n >= N) return;
  float v[10], mx = -3.4e38f;
  #pragma unroll
  for (int c = 0; c < 10; ++c) { v[c] = agg[n * 10 + c] + b2[c]; mx = fmaxf(mx, v[c]); }
  float se = 0.f;
  #pragma unroll
  for (int c = 0; c < 10; ++c) se += __expf(v[c] - mx);
  float lse = __logf(se);
  #pragma unroll
  for (int c = 0; c < 10; ++c) out[n * 10 + c] = v[c] - mx - lse;
}

extern "C" void kernel_launch(void* const* d_in, const int* in_sizes, int n_in,
                              void* d_out, int out_size, void* d_ws, size_t ws_size,
                              hipStream_t stream) {
  const float* x    = (const float*)d_in[0];
  const float* W1   = (const float*)d_in[1];
  const float* as1w = (const float*)d_in[2];
  const float* ad1w = (const float*)d_in[3];
  const float* b1   = (const float*)d_in[4];
  const float* W2   = (const float*)d_in[5];
  const float* as2w = (const float*)d_in[6];
  const float* ad2w = (const float*)d_in[7];
  const float* b2   = (const float*)d_in[8];
  const long long* ei = (const long long*)d_in[9];   // int64 edge_index [2,E]

  const long long N = in_sizes[0] / 512;
  const long long E = in_sizes[9] / 2;
  const long long EE = E + N;

  char* ws = (char*)d_ws;
  size_t off = 0;
  auto alloc = [&](size_t bytes) { void* p = ws + off; off = (off + bytes + 255) & ~(size_t)255; return p; };

  _Float16* w1t  = (_Float16*)alloc(512 * 64 * 2);
  float*    h1   = (float*)   alloc((size_t)N * 64 * 4);
  float*    agg1 = (float*)   alloc((size_t)N * 64 * 4);
  float*    as1  = (float*)   alloc((size_t)N * 8 * 4);
  float*    ad1  = (float*)   alloc((size_t)N * 8 * 4);
  unsigned* m1   = (unsigned*)alloc((size_t)N * 8 * 4);
  float*    den1 = (float*)   alloc((size_t)N * 8 * 4);
  float*    h2   = (float*)   alloc((size_t)N * 10 * 4);
  float*    agg2 = (float*)   alloc((size_t)N * 10 * 4);
  float*    as2  = (float*)   alloc((size_t)N * 4);
  float*    ad2  = (float*)   alloc((size_t)N * 4);
  unsigned* m2   = (unsigned*)alloc((size_t)N * 4);
  float*    den2 = (float*)   alloc((size_t)N * 4);

  auto G = [](long long n) { return (unsigned)((n + 255) / 256); };
  const dim3 B(256);

  k_w1t      <<<G(512 * 64), B, 0, stream>>>(W1, w1t);
  k_gemm1    <<<(unsigned)((N + 127) / 128), B, 0, stream>>>(x, w1t, h1, (int)N);
  k_alpha1   <<<G(N * 8), B, 0, stream>>>(h1, as1w, ad1w, as1, ad1, N);
  k_init1    <<<G(N * 64), B, 0, stream>>>(agg1, den1, m1, N);
  k_edge_max1<<<G(EE * 8), B, 0, stream>>>(ei, E, N, as1, ad1, m1);
  k_edge_sum1<<<G(EE * 8), B, 0, stream>>>(ei, E, N, as1, ad1, m1, den1);
  k_edge_agg1<<<G(EE * 64), B, 0, stream>>>(ei, E, N, as1, ad1, m1, den1, h1, agg1);
  k_bias_elu <<<G(N * 64), B, 0, stream>>>(agg1, b1, N);
  k_gemm2    <<<G(N * 10), B, 0, stream>>>(agg1, W2, h2, N);
  k_alpha2   <<<G(N), B, 0, stream>>>(h2, as2w, ad2w, as2, ad2, N);
  k_init2    <<<G(N * 10), B, 0, stream>>>(agg2, den2, m2, N);
  k_edge_max2<<<G(EE), B, 0, stream>>>(ei, E, N, as2, ad2, m2);
  k_edge_sum2<<<G(EE), B, 0, stream>>>(ei, E, N, as2, ad2, m2, den2);
  k_edge_agg2<<<G(EE), B, 0, stream>>>(ei, E, N, as2, ad2, m2, den2, h2, agg2);
  k_logsm    <<<G(N), B, 0, stream>>>(agg2, b2, (float*)d_out, N);
}
